// SchnetModel_4183298146312
// MI455X (gfx1250) — matrix operand order repaired and checked by reading, not compiled, not run on hardware
//
#include <hip/hip_runtime.h>
#include <hip/hip_bf16.h>

// ---------------------------------------------------------------------------
// SchNet forward on gfx1250 (MI455X).
// All GEMMs via v_wmma_f32_16x16x32_f16; both MLP layers fused through LDS;
// edge message construction + segment_sum fused into the edge-GEMM epilogue
// via global_atomic_add_f32. Block M-tile = 32 rows (2x16 subtiles) so each
// B fragment is reused by two WMMAs -> halves weight/L2 traffic.
// ---------------------------------------------------------------------------

typedef __attribute__((ext_vector_type(16))) _Float16 v16h;
typedef __attribute__((ext_vector_type(8)))  float    v8f;

#define BB   64
#define NN   128
#define EE   2048
#define HH   256
#define LL   6
#define TN   (BB * NN)       // 8192
#define TE   (BB * EE)       // 131072
#define KPAD 64              // padded RBF size (50 -> 64)
#define MT   32              // block M-tile rows (2 x 16)

// shifted softplus: softplus(x) - log(2)
__device__ __forceinline__ float ssp_f(float x) {
  return fmaxf(x, 0.f) + log1pf(__expf(-fabsf(x))) - 0.69314718056f;
}
__device__ __forceinline__ float softplus_f(float x) {
  return fmaxf(x, 0.f) + log1pf(__expf(-fabsf(x)));
}

// ---- WMMA fragment loaders ------------------------------------------------
// A fragment (16x32, f16), documented CDNA5 layout:
//   lane m = lane&15 ; K-half = lane>>4 (adds +8)
//   VGPR j: K = (j&3)*2 + (j>>2)*16 (+ khalf*8), pairs packed per dword
__device__ __forceinline__ v16h load_a_frag(const _Float16* __restrict__ sA,
                                            int ldk, int lane, int k0) {
  const int m  = lane & 15;
  const int kh = lane >> 4;
  const _Float16* p = sA + m * ldk + k0 + kh * 8;
  v16h a;
#pragma unroll
  for (int j = 0; j < 8; ++j) {
    const int off = (j & 3) * 2 + (j >> 2) * 16;
    a[2 * j]     = p[off];
    a[2 * j + 1] = p[off + 1];
  }
  return a;
}

// B fragment (32x16, f16): same K order as A: elems 0-7 K=8*kh+0..7,
// elems 8-15 K=16+8*kh+0..7, column n = lane&15. Weights stored N-major
// (transposed) => two 16B vector loads per lane.
__device__ __forceinline__ v16h load_b_frag(const _Float16* __restrict__ Wt,
                                            int ldk, int n0, int lane, int k0) {
  typedef __attribute__((ext_vector_type(8))) _Float16 v8h_t;
  const int n  = n0 + (lane & 15);
  const int kh = lane >> 4;
  const _Float16* p = Wt + (size_t)n * ldk + k0 + kh * 8;
  const v8h_t lo = *(const v8h_t*)p;
  const v8h_t hi = *(const v8h_t*)(p + 16);
  v16h r;
#pragma unroll
  for (int i = 0; i < 8; ++i) { r[i] = lo[i]; r[8 + i] = hi[i]; }
  return r;
}

__device__ __forceinline__ v8f wmma_f16(v16h a, v16h b, v8f c) {
  return __builtin_amdgcn_wmma_f32_16x16x32_f16(false, a, false, b,
                                                (short)0, c, false, false);
}

// ---------------------------------------------------------------------------
// Fused 2-layer MLP:  OUT = ssp(A @ W1 + b1) @ W2 + b2   (tile M=32, N=256)
// 8 waves: wave w owns columns [32w, 32w+32), both 16-row M subtiles.
// MODE 0: write OUT as f16 (nmsg)
// MODE 1: edge path: OUT=gates; msg = gates * nmsg[src]; atomicAdd msum[dst]
// MODE 2: state path: x += OUT ; refresh x_h
// MODE 3: stage-1 only: write ssp(A@W1+b1) as f16 (readout hidden)
// ---------------------------------------------------------------------------
template <int K1, int MODE, bool AF32>
__global__ __launch_bounds__(256) void mlp_fused_kernel(
    const void* __restrict__ Aptr,        // [M][K1] f32 or f16
    const _Float16* __restrict__ W1t,     // [256][K1]  (N-major)
    const float* __restrict__ b1,         // [256]
    const _Float16* __restrict__ W2t,     // [256][256] (N-major)
    const float* __restrict__ b2,         // [256]
    _Float16* __restrict__ out_h,         // MODE 0 / 3
    float* __restrict__ x,                // MODE 2 (residual in/out)
    _Float16* __restrict__ x_h,           // MODE 2
    const int* __restrict__ edges,        // MODE 1: (B,E,2) flat
    const _Float16* __restrict__ nmsg_h,  // MODE 1
    float* __restrict__ msum)             // MODE 1 (atomic accum)
{
  __shared__ _Float16 sA[MT * 256];
  __shared__ _Float16 sH1[MT * 256];
  __shared__ int s_src[MT];
  __shared__ int s_dst[MT];

  const int tile = blockIdx.x;
  const int tid  = threadIdx.x;

  // ---- preload A tile (MT x K1) into LDS ----
  {
    const int elems = MT * K1;
    if (AF32) {
      const float* A = (const float*)Aptr + (size_t)tile * elems;
      for (int t = tid; t < elems; t += 256) sA[t] = (_Float16)A[t];
    } else {
      const _Float16* A = (const _Float16*)Aptr + (size_t)tile * elems;
      for (int t = tid; t < elems; t += 256) sA[t] = A[t];
    }
  }
  if (MODE == 1 && tid < MT) {
    const int te   = tile * MT + tid;
    const int boff = (te >> 11) << 7;        // (te / E) * N
    s_src[tid] = edges[2 * te]     + boff;
    s_dst[tid] = edges[2 * te + 1] + boff;
  }
  __syncthreads();

  const int lane = tid & 31;
  const int wave = tid >> 5;       // 8 waves, each owns cols [32w, 32w+32)
  const int kh   = lane >> 4;
  const int ln   = lane & 15;
  const int c0   = wave * 32 + ln;
  const int c1   = c0 + 16;

  // acc[2*ms + nt]: M subtile ms in {0,1}, N tile nt in {0,1}
  v8f acc0, acc1, acc2, acc3;

  // ---- stage 1: H1 = ssp(A @ W1 + b1) ----
  {
    const float bv0 = b1[c0];
    const float bv1 = b1[c1];
#pragma unroll
    for (int v = 0; v < 8; ++v) {
      acc0[v] = bv0; acc1[v] = bv1; acc2[v] = bv0; acc3[v] = bv1;
    }
  }
#pragma unroll
  for (int k0 = 0; k0 < K1; k0 += 32) {
    const v16h w0 = load_b_frag(W1t, K1, wave * 32,      lane, k0);
    const v16h w1 = load_b_frag(W1t, K1, wave * 32 + 16, lane, k0);
    const v16h a0 = load_a_frag(sA,            K1, lane, k0);
    const v16h a1 = load_a_frag(sA + 16 * K1,  K1, lane, k0);
    acc0 = wmma_f16(a0, w0, acc0);
    acc1 = wmma_f16(a0, w1, acc1);
    acc2 = wmma_f16(a1, w0, acc2);
    acc3 = wmma_f16(a1, w1, acc3);
  }
#pragma unroll
  for (int v = 0; v < 8; ++v) {
    const int r0 = v + kh * 8;
    const int r1 = r0 + 16;
    sH1[r0 * 256 + c0] = (_Float16)ssp_f(acc0[v]);
    sH1[r0 * 256 + c1] = (_Float16)ssp_f(acc1[v]);
    sH1[r1 * 256 + c0] = (_Float16)ssp_f(acc2[v]);
    sH1[r1 * 256 + c1] = (_Float16)ssp_f(acc3[v]);
  }
  __syncthreads();

  if (MODE == 3) {  // readout stage-1 only
    _Float16* dst = out_h + (size_t)tile * MT * 256;
    for (int t = tid; t < MT * 256; t += 256) dst[t] = sH1[t];
    return;
  }

  // ---- stage 2: OUT = H1 @ W2 + b2 ----
  {
    const float bv0 = b2[c0];
    const float bv1 = b2[c1];
#pragma unroll
    for (int v = 0; v < 8; ++v) {
      acc0[v] = bv0; acc1[v] = bv1; acc2[v] = bv0; acc3[v] = bv1;
    }
  }
#pragma unroll
  for (int k0 = 0; k0 < 256; k0 += 32) {
    const v16h w0 = load_b_frag(W2t, 256, wave * 32,      lane, k0);
    const v16h w1 = load_b_frag(W2t, 256, wave * 32 + 16, lane, k0);
    const v16h a0 = load_a_frag(sH1,             256, lane, k0);
    const v16h a1 = load_a_frag(sH1 + 16 * 256,  256, lane, k0);
    acc0 = wmma_f16(a0, w0, acc0);
    acc1 = wmma_f16(a0, w1, acc1);
    acc2 = wmma_f16(a1, w0, acc2);
    acc3 = wmma_f16(a1, w1, acc3);
  }

  if (MODE == 0) {                      // nmsg (f16)
    _Float16* o = out_h + (size_t)tile * MT * 256;
#pragma unroll
    for (int v = 0; v < 8; ++v) {
      const int r0 = v + kh * 8;
      const int r1 = r0 + 16;
      o[r0 * 256 + c0] = (_Float16)acc0[v];
      o[r0 * 256 + c1] = (_Float16)acc1[v];
      o[r1 * 256 + c0] = (_Float16)acc2[v];
      o[r1 * 256 + c1] = (_Float16)acc3[v];
    }
  } else if (MODE == 1) {               // gates * nmsg[src] -> msum[dst]
#pragma unroll
    for (int v = 0; v < 8; ++v) {
      const int r0 = v + kh * 8;
      const int r1 = r0 + 16;
      {
        const int s = s_src[r0], d = s_dst[r0];
        const float m0 = acc0[v] * (float)nmsg_h[(size_t)s * 256 + c0];
        const float m1 = acc1[v] * (float)nmsg_h[(size_t)s * 256 + c1];
        __hip_atomic_fetch_add(&msum[(size_t)d * 256 + c0], m0,
                               __ATOMIC_RELAXED, __HIP_MEMORY_SCOPE_AGENT);
        __hip_atomic_fetch_add(&msum[(size_t)d * 256 + c1], m1,
                               __ATOMIC_RELAXED, __HIP_MEMORY_SCOPE_AGENT);
      }
      {
        const int s = s_src[r1], d = s_dst[r1];
        const float m0 = acc2[v] * (float)nmsg_h[(size_t)s * 256 + c0];
        const float m1 = acc3[v] * (float)nmsg_h[(size_t)s * 256 + c1];
        __hip_atomic_fetch_add(&msum[(size_t)d * 256 + c0], m0,
                               __ATOMIC_RELAXED, __HIP_MEMORY_SCOPE_AGENT);
        __hip_atomic_fetch_add(&msum[(size_t)d * 256 + c1], m1,
                               __ATOMIC_RELAXED, __HIP_MEMORY_SCOPE_AGENT);
      }
    }
  } else {                              // MODE 2: residual state update
#pragma unroll
    for (int v = 0; v < 8; ++v) {
      const int r0 = v + kh * 8;
      const int r1 = r0 + 16;
      const size_t base0 = ((size_t)tile * MT + r0) * 256;
      const size_t base1 = ((size_t)tile * MT + r1) * 256;
      const float n0 = x[base0 + c0] + acc0[v];
      const float n1 = x[base0 + c1] + acc1[v];
      const float n2 = x[base1 + c0] + acc2[v];
      const float n3 = x[base1 + c1] + acc3[v];
      x[base0 + c0] = n0;  x_h[base0 + c0] = (_Float16)n0;
      x[base0 + c1] = n1;  x_h[base0 + c1] = (_Float16)n1;
      x[base1 + c0] = n2;  x_h[base1 + c0] = (_Float16)n2;
      x[base1 + c1] = n3;  x_h[base1 + c1] = (_Float16)n3;
    }
  }
}

// ---------------------------------------------------------------------------
// small helper kernels
// ---------------------------------------------------------------------------
__global__ void convw_kernel(const float* __restrict__ src,  // [K][256]
                             _Float16* __restrict__ dst,     // [256][Kpad]
                             int K, int Kpad) {
  const int idx = blockIdx.x * blockDim.x + threadIdx.x;
  if (idx >= 256 * Kpad) return;
  const int n = idx / Kpad;
  const int k = idx % Kpad;
  dst[idx] = (k < K) ? (_Float16)src[k * 256 + n] : (_Float16)0.f;
}

__global__ void embed_kernel(const int* __restrict__ nodes,
                             const float* __restrict__ embed,
                             float* __restrict__ x, _Float16* __restrict__ x_h) {
  const int idx = blockIdx.x * blockDim.x + threadIdx.x;
  if (idx >= TN * HH) return;
  const int i = idx >> 8;
  const int h = idx & 255;
  const float v = embed[nodes[i] * HH + h];
  x[idx]   = v;
  x_h[idx] = (_Float16)v;
}

__global__ void rbf_kernel(const float* __restrict__ ef,    // [TE]
                           _Float16* __restrict__ e_h) {    // [TE][64]
  const int idx = blockIdx.x * blockDim.x + threadIdx.x;
  if (idx >= TE * KPAD) return;
  const int te = idx >> 6;
  const int k  = idx & 63;
  float v = 0.f;
  if (k < 50) {
    const float t = ef[te] - 0.1f * (float)k;
    v = __expf(-t * t * 50.f);   // 1/(2*0.1^2) = 50
  }
  e_h[idx] = (_Float16)v;
}

__global__ void zero_kernel(float4* __restrict__ p, int n4) {
  const int idx = blockIdx.x * blockDim.x + threadIdx.x;
  if (idx < n4) p[idx] = make_float4(0.f, 0.f, 0.f, 0.f);
}

// per-batch readout: y[n] = dot(hidden[n], ro_w2) + ro_b2; g = sum_n y[n];
// g = g*STD + MEAN*N ; z = g*ev_w + ev_b ; head transforms
__global__ __launch_bounds__(256) void readout_kernel(
    const _Float16* __restrict__ hidden,  // [TN][256]
    const float* __restrict__ ro_w2,      // [256]
    const float* __restrict__ ro_b2,      // [1]
    const float* __restrict__ ev_w,       // [4]
    const float* __restrict__ ev_b,       // [4]
    float* __restrict__ out)              // [B][4]
{
  __shared__ float red[256];
  const int b = blockIdx.x;
  const int h = threadIdx.x;
  float cs = 0.f;
  const _Float16* base = hidden + (size_t)b * NN * HH + h;
#pragma unroll 4
  for (int n = 0; n < NN; ++n) cs += (float)base[(size_t)n * HH];
  red[h] = cs * ro_w2[h];
  __syncthreads();
  for (int s = 128; s > 0; s >>= 1) {
    if (h < s) red[h] += red[h + s];
    __syncthreads();
  }
  if (h == 0) {
    const float y = red[0] + (float)NN * ro_b2[0];
    const float g = y * 2.0f + (-1.5f) * (float)NN;   // STD, MEAN
#pragma unroll
    for (int k = 0; k < 4; ++k) {
      const float z  = g * ev_w[k] + ev_b[k];
      const float sp = softplus_f(z);
      out[b * 4 + k] = (k == 0) ? z : ((k == 2) ? sp + 1.f : sp);
    }
  }
}

// ---------------------------------------------------------------------------
extern "C" void kernel_launch(void* const* d_in, const int* in_sizes, int n_in,
                              void* d_out, int out_size, void* d_ws, size_t ws_size,
                              hipStream_t stream) {
  (void)in_sizes; (void)n_in; (void)out_size; (void)ws_size;

  const int*   nodes  = (const int*)  d_in[0];
  const int*   edges  = (const int*)  d_in[2];
  const float* efeat  = (const float*)d_in[3];
  const float* embed  = (const float*)d_in[5];
  const float* me_w1  = (const float*)d_in[6];
  const float* me_b1  = (const float*)d_in[7];
  const float* me_w2  = (const float*)d_in[8];
  const float* me_b2  = (const float*)d_in[9];
  const float* mn_w1  = (const float*)d_in[10];
  const float* mn_b1  = (const float*)d_in[11];
  const float* mn_w2  = (const float*)d_in[12];
  const float* mn_b2  = (const float*)d_in[13];
  const float* st_w1  = (const float*)d_in[14];
  const float* st_b1  = (const float*)d_in[15];
  const float* st_w2  = (const float*)d_in[16];
  const float* st_b2  = (const float*)d_in[17];
  const float* ro_w1  = (const float*)d_in[18];
  const float* ro_b1  = (const float*)d_in[19];
  const float* ro_w2  = (const float*)d_in[20];
  const float* ro_b2  = (const float*)d_in[21];
  const float* ev_w   = (const float*)d_in[22];
  const float* ev_b   = (const float*)d_in[23];

  // ---- workspace layout ----
  char* ws = (char*)d_ws;
  size_t off = 0;
  auto take = [&](size_t bytes) -> char* {
    char* p = ws + off;
    off += (bytes + 255) & ~(size_t)255;
    return p;
  };
  _Float16* me_w1_h = (_Float16*)take((size_t)LL * 256 * KPAD * 2);
  _Float16* me_w2_h = (_Float16*)take((size_t)LL * 256 * 256 * 2);
  _Float16* mn_w1_h = (_Float16*)take((size_t)LL * 256 * 256 * 2);
  _Float16* mn_w2_h = (_Float16*)take((size_t)LL * 256 * 256 * 2);
  _Float16* st_w1_h = (_Float16*)take((size_t)LL * 256 * 256 * 2);
  _Float16* st_w2_h = (_Float16*)take((size_t)LL * 256 * 256 * 2);
  _Float16* ro_w1_h = (_Float16*)take((size_t)256 * 256 * 2);
  float*    x       = (float*)   take((size_t)TN * HH * 4);
  _Float16* x_h     = (_Float16*)take((size_t)TN * HH * 2);
  _Float16* nmsg_h  = (_Float16*)take((size_t)TN * HH * 2);   // reused as hidden
  float*    msum    = (float*)   take((size_t)TN * HH * 4);
  _Float16* e_h     = (_Float16*)take((size_t)TE * KPAD * 2);

  // ---- weight conversion (f32 row-major -> f16 N-major, K padded) ----
  for (int l = 0; l < LL; ++l) {
    convw_kernel<<<(256 * KPAD) / 256, 256, 0, stream>>>(
        me_w1 + (size_t)l * 50 * 256, me_w1_h + (size_t)l * 256 * KPAD, 50, KPAD);
    convw_kernel<<<256, 256, 0, stream>>>(
        me_w2 + (size_t)l * 65536, me_w2_h + (size_t)l * 65536, 256, 256);
    convw_kernel<<<256, 256, 0, stream>>>(
        mn_w1 + (size_t)l * 65536, mn_w1_h + (size_t)l * 65536, 256, 256);
    convw_kernel<<<256, 256, 0, stream>>>(
        mn_w2 + (size_t)l * 65536, mn_w2_h + (size_t)l * 65536, 256, 256);
    convw_kernel<<<256, 256, 0, stream>>>(
        st_w1 + (size_t)l * 65536, st_w1_h + (size_t)l * 65536, 256, 256);
    convw_kernel<<<256, 256, 0, stream>>>(
        st_w2 + (size_t)l * 65536, st_w2_h + (size_t)l * 65536, 256, 256);
  }
  convw_kernel<<<256, 256, 0, stream>>>(ro_w1, ro_w1_h, 256, 256);

  // ---- node embedding + RBF features ----
  embed_kernel<<<(TN * HH) / 256, 256, 0, stream>>>(nodes, embed, x, x_h);
  rbf_kernel<<<(TE * KPAD) / 256, 256, 0, stream>>>(efeat, e_h);

  // ---- interaction layers ----
  for (int l = 0; l < LL; ++l) {
    // nmsg = ssp(x @ mn_w1 + b1) @ mn_w2 + b2
    mlp_fused_kernel<256, 0, false><<<TN / MT, 256, 0, stream>>>(
        x_h, mn_w1_h + (size_t)l * 65536, mn_b1 + l * 256,
        mn_w2_h + (size_t)l * 65536, mn_b2 + l * 256,
        nmsg_h, nullptr, nullptr, nullptr, nullptr, nullptr);
    // msum = 0
    zero_kernel<<<(TN * HH / 4) / 256, 256, 0, stream>>>((float4*)msum, TN * HH / 4);
    // gates = ssp(e @ me_w1 + b1) @ me_w2 + b2 ; scatter nmsg[src]*gates -> msum[dst]
    mlp_fused_kernel<KPAD, 1, false><<<TE / MT, 256, 0, stream>>>(
        e_h, me_w1_h + (size_t)l * 256 * KPAD, me_b1 + l * 256,
        me_w2_h + (size_t)l * 65536, me_b2 + l * 256,
        nullptr, nullptr, nullptr, edges, nmsg_h, msum);
    // x += ssp(msum @ st_w1 + b1) @ st_w2 + b2
    mlp_fused_kernel<256, 2, true><<<TN / MT, 256, 0, stream>>>(
        msum, st_w1_h + (size_t)l * 65536, st_b1 + l * 256,
        st_w2_h + (size_t)l * 65536, st_b2 + l * 256,
        nullptr, x, x_h, nullptr, nullptr, nullptr);
  }

  // ---- readout ----
  mlp_fused_kernel<256, 3, false><<<TN / MT, 256, 0, stream>>>(
      x_h, ro_w1_h, ro_b1, nullptr, nullptr,
      nmsg_h /* hidden */, nullptr, nullptr, nullptr, nullptr, nullptr);
  readout_kernel<<<BB, 256, 0, stream>>>(nmsg_h, ro_w2, ro_b2, ev_w, ev_b,
                                         (float*)d_out);
}
